// MTL_Decoder_28132035789067
// MI455X (gfx1250) — matrix operand order repaired and checked by reading, not compiled, not run on hardware
//
#include <hip/hip_runtime.h>
#include <math.h>

#define B_  64
#define P_  128
#define N1_ 1001
#define E_  128
#define H_  8
#define D_  16

typedef __attribute__((ext_vector_type(16))) __bf16 v16bf;
typedef __attribute__((ext_vector_type(8)))  __bf16 v8bf;
typedef __attribute__((ext_vector_type(8)))  float  v8f;

static __device__ __forceinline__ __bf16 f2bf(float f) { return (__bf16)f; }

static __device__ __forceinline__ v16bf ld16g(const __bf16* p) { return *(const v16bf*)p; }
static __device__ __forceinline__ v16bf cat8(v8bf lo, v8bf hi) {
  return __builtin_shufflevector(lo, hi, 0, 1, 2, 3, 4, 5, 6, 7, 8, 9, 10, 11, 12, 13, 14, 15);
}
// A-fragment: 16 bf16 (8 at base, 8 at base+16 elements), 16B-aligned
static __device__ __forceinline__ v16bf ldfragA(const __bf16* p) {
  v8bf lo = *(const v8bf*)p;
  v8bf hi = *(const v8bf*)(p + 16);
  return cat8(lo, hi);
}

static __device__ __forceinline__ v8f wmma_bf(v16bf a, v16bf b, v8f c) {
  return __builtin_amdgcn_wmma_f32_16x16x32_bf16(false, a, false, b, (short)0, c, false, false);
}

// ---------------------------------------------------------------- 1a: enc f32 -> bf16
__global__ void k_cvt_enc(const float* __restrict__ src, __bf16* __restrict__ dst, int n) {
  int i = blockIdx.x * 256 + threadIdx.x;
  if (i < n) dst[i] = f2bf(src[i]);
}

// ---------------------------------------------------------------- 1b: weights -> bf16, transposed
// WT[n*128+k] = W[k*128+n], 4 matrices of 128x128 (Wq: first 128 rows only).
__global__ void k_prep_w(const float* __restrict__ Wq, const float* __restrict__ Wk,
                         const float* __restrict__ Wv, const float* __restrict__ Wc,
                         __bf16* __restrict__ WqT, __bf16* __restrict__ WkT,
                         __bf16* __restrict__ WvT, __bf16* __restrict__ WcT) {
  int i = blockIdx.x * 256 + threadIdx.x;     // 0 .. 4*16384-1
  int mat = i >> 14, rem = i & 16383;
  int n = rem >> 7, k = rem & 127;
  const float* src = (mat == 0) ? Wq : (mat == 1) ? Wk : (mat == 2) ? Wv : Wc;
  __bf16* dst = (mat == 0) ? WqT : (mat == 1) ? WkT : (mat == 2) ? WvT : WcT;
  dst[n * 128 + k] = f2bf(src[k * 128 + n]);
}

// ---------------------------------------------------------------- 2: K = X@Wk, V = X@Wv
// grid (63, B), block 128 (4 waves). Each wave: 16x32 tile of K and of V.
__global__ void k_kv(const float* __restrict__ enc, const __bf16* __restrict__ WkT,
                     const __bf16* __restrict__ WvT, __bf16* __restrict__ kbf,
                     __bf16* __restrict__ vbf) {
  __shared__ __bf16 xs[16 * 128];
  const int b = blockIdx.y, row0 = blockIdx.x * 16, tid = threadIdx.x;
  for (int i = tid; i < 16 * 128; i += 128) {
    int r = i >> 7, c = i & 127, gr = row0 + r;
    xs[i] = (gr < N1_) ? f2bf(enc[((size_t)b * N1_ + gr) * E_ + c]) : f2bf(0.f);
  }
  __syncthreads();
  const int wave = tid >> 5, lane = tid & 31, hf = lane >> 4, m = lane & 15;
  const int n0 = wave * 32;
  v8f ck0{}, ck1{}, cv0{}, cv1{};
#pragma unroll
  for (int kk = 0; kk < 128; kk += 32) {
    v16bf a = ldfragA(&xs[m * 128 + kk + 8 * hf]);
    const int kbB = kk + 8 * hf;
    v16bf bk0 = ldfragA(&WkT[(n0 + m) * 128 + kbB]);
    v16bf bk1 = ldfragA(&WkT[(n0 + 16 + m) * 128 + kbB]);
    v16bf bv0 = ldfragA(&WvT[(n0 + m) * 128 + kbB]);
    v16bf bv1 = ldfragA(&WvT[(n0 + 16 + m) * 128 + kbB]);
    ck0 = wmma_bf(a, bk0, ck0);
    ck1 = wmma_bf(a, bk1, ck1);
    cv0 = wmma_bf(a, bv0, cv0);
    cv1 = wmma_bf(a, bv1, cv1);
  }
#pragma unroll
  for (int r = 0; r < 8; ++r) {
    const int gr = row0 + r + 8 * hf;
    if (gr < N1_) {
      const size_t base = ((size_t)b * N1_ + gr) * 128;
      kbf[base + n0 + m]      = f2bf(ck0[r]);
      kbf[base + n0 + 16 + m] = f2bf(ck1[r]);
      vbf[base + n0 + m]      = f2bf(cv0[r]);
      vbf[base + n0 + 16 + m] = f2bf(cv1[r]);
    }
  }
}

// ---------------------------------------------------------------- 3: Q = [X|attr]@Wq
// grid 512 (16-row tiles of B*P rows), block 128.
__global__ void k_q(const float* __restrict__ xlast, const float* __restrict__ attr,
                    const float* __restrict__ Wq, const __bf16* __restrict__ WqT,
                    __bf16* __restrict__ qbf) {
  __shared__ __bf16 xs[16 * 136];               // padded row stride for 16B alignment
  __shared__ float  at[16 * 4];
  const int row0 = blockIdx.x * 16, tid = threadIdx.x;
  for (int i = tid; i < 16 * 128; i += 128) {
    int r = i >> 7, c = i & 127;
    xs[r * 136 + c] = f2bf(xlast[(size_t)(row0 + r) * 128 + c]);
  }
  if (tid < 64) at[tid] = attr[(size_t)(row0 + (tid >> 2)) * 4 + (tid & 3)];
  __syncthreads();
  const int wave = tid >> 5, lane = tid & 31, hf = lane >> 4, m = lane & 15;
  const int n0 = wave * 32;
  v8f c0{}, c1{};
#pragma unroll
  for (int kk = 0; kk < 128; kk += 32) {
    v16bf a = ldfragA(&xs[m * 136 + kk + 8 * hf]);
    const int kbB = kk + 8 * hf;
    v16bf b0 = ldfragA(&WqT[(n0 + m) * 128 + kbB]);
    v16bf b1 = ldfragA(&WqT[(n0 + 16 + m) * 128 + kbB]);
    c0 = wmma_bf(a, b0, c0);
    c1 = wmma_bf(a, b1, c1);
  }
  float wq0[4], wq1[4];
#pragma unroll
  for (int j = 0; j < 4; ++j) {                 // tail weights K = 128..131 (f32)
    wq0[j] = Wq[(128 + j) * 128 + n0 + m];
    wq1[j] = Wq[(128 + j) * 128 + n0 + 16 + m];
  }
#pragma unroll
  for (int r = 0; r < 8; ++r) {
    const int rr = r + 8 * hf;
    float s0 = c0[r], s1 = c1[r];
#pragma unroll
    for (int j = 0; j < 4; ++j) {
      const float xv = at[rr * 4 + j];
      s0 += xv * wq0[j];
      s1 += xv * wq1[j];
    }
    qbf[(size_t)(row0 + rr) * 128 + n0 + m]      = f2bf(s0);
    qbf[(size_t)(row0 + rr) * 128 + n0 + 16 + m] = f2bf(s1);
  }
}

// ---------------------------------------------------------------- 4: flash attention
// grid (8, H, B), block 32 (one wave per 16-query tile of one head).
__global__ void k_attn(const __bf16* __restrict__ qbf, const __bf16* __restrict__ kbf,
                       const __bf16* __restrict__ vbf, const float* __restrict__ mask,
                       __bf16* __restrict__ obf) {
  __shared__ float tr[16 * 16];
  const int pt = blockIdx.x, h = blockIdx.y, b = blockIdx.z;
  const int lane = threadIdx.x, hf = lane >> 4, m = lane & 15;

  v16bf qa{};                                   // A: 16 rows x K(=d, padded 16->32)
  {
    v8bf lo = *(const v8bf*)(qbf + ((size_t)b * P_ + pt * 16 + m) * 128 + h * 16 + 8 * hf);
    v8bf zz{};
    qa = cat8(lo, zz);
  }
  v8f acc{};
  float mrun[8], lrun[8];
#pragma unroll
  for (int r = 0; r < 8; ++r) { mrun[r] = -__builtin_inff(); lrun[r] = 0.f; }

  for (int nt = 0; nt < 63; ++nt) {
    const int nb = nt * 16;
    const int n = nb + m;
    const bool nvalid = n < N1_;
    v16bf kf{};                                 // B: K(=d) x 16 cols(n)
    if (nvalid) kf = cat8(*(const v8bf*)&kbf[((size_t)b * N1_ + n) * 128 + h * 16 + 8 * hf], v8bf{});
    v8f S{};
    S = wmma_bf(qa, kf, S);

    float sv[8];
#pragma unroll
    for (int r = 0; r < 8; ++r) {
      const int p = pt * 16 + r + 8 * hf;
      sv[r] = nvalid ? (S[r] * 0.25f + mask[((size_t)b * P_ + p) * N1_ + n])
                     : -__builtin_inff();
    }
    float pv[8];
#pragma unroll
    for (int r = 0; r < 8; ++r) {               // online softmax per row (16-lane half)
      float x = sv[r];
      for (int off = 1; off < 16; off <<= 1) x = fmaxf(x, __shfl_xor(x, off));
      const float mnew = fmaxf(mrun[r], x);
      const float mc = fmaxf(mnew, -1e30f);     // NaN-free when whole row masked
      const float scale = __expf(mrun[r] - mc);
      pv[r] = __expf(sv[r] - mc);
      float rs = pv[r];
      for (int off = 1; off < 16; off <<= 1) rs += __shfl_xor(rs, off);
      lrun[r] = lrun[r] * scale + rs;
      acc[r] *= scale;
      mrun[r] = mnew;
    }
    // transpose probabilities (C layout -> A layout) through LDS
#pragma unroll
    for (int r = 0; r < 8; ++r) tr[(r + 8 * hf) * 16 + m] = pv[r];
    __syncthreads();
    v16bf wa{};
#pragma unroll
    for (int j = 0; j < 8; ++j) wa[j] = f2bf(tr[m * 16 + 8 * hf + j]);
    __syncthreads();
    v16bf vf{};                                 // B: K(=n, padded 16->32) x 16 cols(d)
    {
      const __bf16* vb = vbf + ((size_t)b * N1_ + nb + 8 * hf) * 128 + h * 16 + m;
      if (nt < 62) {
#pragma unroll
        for (int j = 0; j < 8; ++j) vf[j] = vb[j * 128];
      } else {
#pragma unroll
        for (int j = 0; j < 8; ++j)
          if (nb + 8 * hf + j < N1_) vf[j] = vb[j * 128];
      }
    }
    acc = wmma_bf(wa, vf, acc);
  }
#pragma unroll
  for (int r = 0; r < 8; ++r) {
    const float o = acc[r] / lrun[r];
    obf[((size_t)b * P_ + pt * 16 + r + 8 * hf) * 128 + h * 16 + m] = f2bf(o);
  }
}

// ---------------------------------------------------------------- 5: mh = out@Wc + bc
// grid 512, block 128.
__global__ void k_oproj(const __bf16* __restrict__ obf, const __bf16* __restrict__ WcT,
                        const float* __restrict__ bc, __bf16* __restrict__ mhbf) {
  __shared__ __bf16 xs[16 * 128];
  const int row0 = blockIdx.x * 16, tid = threadIdx.x;
  {
    const unsigned* src = (const unsigned*)(obf + (size_t)row0 * 128);
    unsigned* dst = (unsigned*)xs;
    for (int i = tid; i < 1024; i += 128) dst[i] = src[i];
  }
  __syncthreads();
  const int wave = tid >> 5, lane = tid & 31, hf = lane >> 4, m = lane & 15;
  const int n0 = wave * 32;
  v8f c0{}, c1{};
#pragma unroll
  for (int kk = 0; kk < 128; kk += 32) {
    v16bf a = ldfragA(&xs[m * 128 + kk + 8 * hf]);
    const int kbB = kk + 8 * hf;
    v16bf b0 = ldfragA(&WcT[(n0 + m) * 128 + kbB]);
    v16bf b1 = ldfragA(&WcT[(n0 + 16 + m) * 128 + kbB]);
    c0 = wmma_bf(a, b0, c0);
    c1 = wmma_bf(a, b1, c1);
  }
  const float bc0 = bc[n0 + m], bc1 = bc[n0 + 16 + m];
#pragma unroll
  for (int r = 0; r < 8; ++r) {
    mhbf[(size_t)(row0 + r + 8 * hf) * 128 + n0 + m]      = f2bf(c0[r] + bc0);
    mhbf[(size_t)(row0 + r + 8 * hf) * 128 + n0 + 16 + m] = f2bf(c1[r] + bc1);
  }
}

// ---------------------------------------------------------------- 6: logits + tanh-clip + softmax
// grid (8, B), block 128 (4 waves). LDS holds 16 rows x 1008 cols of scores.
__global__ void k_score2(const __bf16* __restrict__ mhbf, const __bf16* __restrict__ encbf,
                         const float* __restrict__ bias, const float* __restrict__ bias1,
                         const float* __restrict__ mask, float* __restrict__ out) {
  __shared__ float srow[16 * 1008];
  const int pt = blockIdx.x, b = blockIdx.y;
  const int tid = threadIdx.x, wave = tid >> 5, lane = tid & 31, hf = lane >> 4, m = lane & 15;
  const int p0 = pt * 16;

  v16bf A[4];
#pragma unroll
  for (int ks = 0; ks < 4; ++ks)
    A[ks] = ldfragA(mhbf + ((size_t)b * P_ + p0 + m) * 128 + ks * 32 + 8 * hf);

  const float invsq = 0.08838834764831845f;     // 1/sqrt(128)

  for (int nt = wave; nt < 63; nt += 4) {
    const int nb = nt * 16;
    const int n = nb + m;
    const bool nvalid = n < N1_;
    v8f S{};
#pragma unroll
    for (int ks = 0; ks < 4; ++ks) {
      v16bf Bf{};
      if (nvalid) Bf = ldfragA(&encbf[((size_t)b * N1_ + n) * 128 + ks * 32 + 8 * hf]);
      S = wmma_bf(A[ks], Bf, S);
    }
#pragma unroll
    for (int r = 0; r < 8; ++r) {
      const int p = p0 + r + 8 * hf;
      float v;
      if (nvalid) {
        const size_t idx = ((size_t)b * P_ + p) * N1_ + n;
        const float t = S[r] + bias[idx] + bias1[idx];
        v = 10.f * tanhf(t * invsq) + mask[idx];
      } else {
        v = -__builtin_inff();
      }
      srow[(r + 8 * hf) * 1008 + nb + m] = v;
    }
  }
  __syncthreads();
  // row softmax: wave w handles rows 4w..4w+3
  for (int r = wave * 4; r < wave * 4 + 4; ++r) {
    float mx = -__builtin_inff();
    for (int i = lane; i < 1008; i += 32) mx = fmaxf(mx, srow[r * 1008 + i]);
    for (int off = 1; off < 32; off <<= 1) mx = fmaxf(mx, __shfl_xor(mx, off));
    float sum = 0.f;
    for (int i = lane; i < 1008; i += 32) {
      const float e = __expf(srow[r * 1008 + i] - mx);
      srow[r * 1008 + i] = e;
      sum += e;
    }
    for (int off = 1; off < 32; off <<= 1) sum += __shfl_xor(sum, off);
    const float inv = 1.f / sum;
    const int p = p0 + r;
    for (int i = lane; i < N1_; i += 32)
      out[((size_t)b * P_ + p) * N1_ + i] = srow[r * 1008 + i] * inv;
  }
}

// ----------------------------------------------------------------------------
extern "C" void kernel_launch(void* const* d_in, const int* in_sizes, int n_in,
                              void* d_out, int out_size, void* d_ws, size_t ws_size,
                              hipStream_t stream) {
  const float* xlast = (const float*)d_in[0];   // (B,P,E)
  const float* attr  = (const float*)d_in[1];   // (B,P,4)
  const float* nmask = (const float*)d_in[2];   // (B,P,N1)
  const float* bias  = (const float*)d_in[3];   // (B,P,N1)
  const float* bias1 = (const float*)d_in[4];   // (B,P,N1)
  const float* enc   = (const float*)d_in[5];   // (B,N1,E)
  const float* Wq    = (const float*)d_in[6];   // (E+4, H*D)
  const float* Wk    = (const float*)d_in[7];   // (E, H*D)
  const float* Wv    = (const float*)d_in[8];   // (E, H*D)
  const float* Wc    = (const float*)d_in[9];   // (H*D, E)
  const float* bc    = (const float*)d_in[10];  // (E,)
  float* out = (float*)d_out;

  char* w = (char*)d_ws;
  const size_t encN = (size_t)B_ * N1_ * 128;   // bf16 elems
  const size_t rowN = (size_t)B_ * P_ * 128;
  const size_t wN   = (size_t)128 * 128;
  __bf16* encbf = (__bf16*)w;            w += encN * 2;
  __bf16* kbf   = (__bf16*)w;            w += encN * 2;
  __bf16* vbf   = (__bf16*)w;            w += encN * 2;
  __bf16* qbf   = (__bf16*)w;            w += rowN * 2;
  __bf16* obf   = (__bf16*)w;            w += rowN * 2;
  __bf16* mhbf  = (__bf16*)w;            w += rowN * 2;
  __bf16* WqT   = (__bf16*)w;            w += wN * 2;
  __bf16* WkT   = (__bf16*)w;            w += wN * 2;
  __bf16* WvT   = (__bf16*)w;            w += wN * 2;
  __bf16* WcT   = (__bf16*)w;            w += wN * 2;

  const int ne = (int)encN;
  k_cvt_enc<<<(ne + 255) / 256, 256, 0, stream>>>(enc, encbf, ne);
  k_prep_w<<<(4 * 16384) / 256, 256, 0, stream>>>(Wq, Wk, Wv, Wc, WqT, WkT, WvT, WcT);
  k_kv<<<dim3(63, B_), 128, 0, stream>>>(enc, WkT, WvT, kbf, vbf);
  k_q<<<dim3((B_ * P_) / 16), 128, 0, stream>>>(xlast, attr, Wq, WqT, qbf);
  k_attn<<<dim3(8, H_, B_), 32, 0, stream>>>(qbf, kbf, vbf, nmask, obf);
  k_oproj<<<dim3((B_ * P_) / 16), 128, 0, stream>>>(obf, WcT, bc, mhbf);
  k_score2<<<dim3(8, B_), 128, 0, stream>>>(mhbf, encbf, bias, bias1, nmask, out);
}
